// PGP_31421980737811
// MI455X (gfx1250) — compile-verified, matrix operand order checked by reading, not run on hardware
//
#include <hip/hip_runtime.h>
#include <math.h>

// ---------------------------------------------------------------------------
// Problem constants (match the reference)
// ---------------------------------------------------------------------------
constexpr int BB     = 32;
constexpr int NN     = 256;
constexpr int NBRS   = 16;
constexpr int HOR    = 16;
constexpr int DNODE  = 128;
constexpr int DTGT   = 128;
constexpr int EMB    = 256;
constexpr int NSAMP  = 64;
constexpr float NEGV = -1.0e9f;

constexpr int AGG_ELEMS = BB * NSAMP * (DTGT + EMB);   // 786432
// log_pi elems = BB*NN*17 = 139264

typedef __attribute__((ext_vector_type(16))) _Float16 v16h;
typedef __attribute__((ext_vector_type(8)))  float    v8f;

// ---------------------------------------------------------------------------
// CDNA5 16-bit fragment index helpers (per CDNA5 ISA layouts).
// A 16x32 fragment: lane = row%16 + 16*h ; element i covers K per:
//   kin<8 : h=0,i=kin ; kin<16: h=1,i=kin-8 ; kin<24: h=0,i=kin-8 ; else h=1,i=kin-16
// We store fragments contiguously as [..][lane][16 halves] so a lane's whole
// fragment is one 32B load (2x b128).
// ---------------------------------------------------------------------------
__device__ __forceinline__ void frag_pos(int kin, int& h, int& i) {
    if      (kin < 8)  { h = 0; i = kin;      }
    else if (kin < 16) { h = 1; i = kin - 8;  }
    else if (kin < 24) { h = 0; i = kin - 8;  }
    else               { h = 1; i = kin - 16; }
}

__device__ __forceinline__ void frag_store(_Float16* fbuf, int KB, int mtile,
                                           int rfr, int k, float v) {
    int kb = k >> 5, kin = k & 31, h, i;
    frag_pos(kin, h, i);
    int lane = rfr + (h << 4);
    fbuf[(((mtile * KB + kb) << 5) + lane) * 16 + i] = (_Float16)v;
}

// ---------------------------------------------------------------------------
// Prep: rewrite a row-major fp32 weight [Kreal x 512] into f16 B-fragment
// layout [KB][32 ntile][32 lane][16], zero-padding K up to KB*32.
// Lane n-column duality: lane holds column ntile*16 + (lane&15), element i
// holds K = kb*32 + ((i<8?i:i+8) + 8*(lane>>4)).
// ---------------------------------------------------------------------------
__global__ __launch_bounds__(256) void prep_w_kernel(const float* __restrict__ W,
                                                     _Float16* __restrict__ frag,
                                                     int KB, int Kreal) {
    int idx = blockIdx.x * 256 + threadIdx.x;
    int total = KB << 14;           // KB*32*32*16
    if (idx >= total) return;
    int i     = idx & 15;
    int lane  = (idx >> 4) & 31;
    int ntile = (idx >> 9) & 31;
    int kb    = idx >> 14;
    int h     = lane >> 4;
    int k     = kb * 32 + ((i < 8 ? i : i + 8) + (h << 3));
    int n     = ntile * 16 + (lane & 15);
    float v   = (k < Kreal) ? W[k * 512 + n] : 0.0f;
    frag[idx] = (_Float16)v;
}

// ---------------------------------------------------------------------------
// One GEMM layer over fragment-resident data.
//   sSrc : LDS A-fragments [4 mtile][KBin][32][16]
//   fragW: global B-fragments [KBin][32 ntile][32][16]
//   Each of 8 waves owns (mtile = w>>1, colgrp = w&1) -> 16x256 of the 64x512
//   output tile; nt loop covers 16 column tiles, kb loop accumulates K.
//   Epilogue: bias + leaky-ReLU; optionally re-fragment into sDst (next A),
//   optionally accumulate layer-3 dot (out = act . w3) into sacc[8].
// ---------------------------------------------------------------------------
template <int KBin, bool STORE, bool SACC>
__device__ __forceinline__ void gemm_layer(const _Float16* __restrict__ sSrc,
                                           const _Float16* __restrict__ fragW,
                                           const float* __restrict__ bias,
                                           _Float16* sDst,
                                           const float* __restrict__ w3,
                                           float* sacc,
                                           int mtile, int colgrp, int lane) {
    const int n = lane & 15;
    const int h = lane >> 4;
    for (int nt = 0; nt < 16; ++nt) {
        const int c0  = colgrp * 256 + nt * 16;
        const int ntg = c0 >> 4;
        v8f acc = {};
#pragma unroll
        for (int kb = 0; kb < KBin; ++kb) {
            v16h a  = *(const v16h*)(sSrc + (((mtile * KBin + kb) << 5) + lane) * 16);
            v16h bw = *(const v16h*)(fragW + (((kb << 5) + ntg) * 32 + lane) * 16);
            acc = __builtin_amdgcn_wmma_f32_16x16x32_f16(
                false, a, false, bw, (short)0, acc, false, false);
        }
        const int   col = c0 + n;
        const float bv  = bias[col];
        float w3v = 0.0f;
        if (SACC) w3v = w3[col];
#pragma unroll
        for (int r = 0; r < 8; ++r) {
            float v = acc[r] + bv;
            v = (v >= 0.0f) ? v : 0.01f * v;          // leaky relu
            if (STORE) {
                int rfr = r + (h << 3);               // row within 16-row tile
                frag_store(sDst, 16, mtile, rfr, col, v);
            }
            if (SACC) sacc[r] += v * w3v;
        }
    }
}

// ---------------------------------------------------------------------------
// Fused 3-layer MLP. 64 rows per block, 256 threads (8 wave32s).
//   EDGE=true : rows are (b,n,nbr); input = [tgt|src|dst|edge_onehot], K=416pad
//   EDGE=false: rows are (b,n);     input = [tgt|node],              K=256
// Layer1 (WMMA) -> LDS frags, Layer2 (WMMA) + Layer3 folded into epilogue.
// ---------------------------------------------------------------------------
template <int KB1, bool EDGE>
__global__ __launch_bounds__(256) void mlp3_kernel(
    const float* __restrict__ tgt, const float* __restrict__ nodes,
    const int* __restrict__ s_next, const int* __restrict__ edge_type,
    const _Float16* __restrict__ fragW1, const _Float16* __restrict__ fragW2,
    const float* __restrict__ b1, const float* __restrict__ b2,
    const float* __restrict__ w3, const float* __restrict__ b3,
    float* __restrict__ out_scores) {

    __shared__ __align__(32) _Float16 sA[4 * KB1 * 32 * 16];   // input A-frags
    __shared__ __align__(32) _Float16 sH1[4 * 16 * 32 * 16];   // hidden1 A-frags
    __shared__ float sScore[64];

    const int t = threadIdx.x;

    // ---- build input rows directly in A-fragment layout -------------------
    {
        const int rl   = t >> 2;                 // local row 0..63
        const int part = t & 3;                  // quarter of the K range
        const int mtile = rl >> 4, rfr = rl & 15;
        const int g = blockIdx.x * 64 + rl;      // global row id
        const float* p0;                         // tgt slice
        const float* p1;                         // src/node slice
        const float* p2 = nullptr;               // dst slice (edge only)
        int et = 0;
        if (EDGE) {
            int b   = g / (NN * NBRS);
            int rem = g - b * NN * NBRS;
            int n   = rem / NBRS;
            int nbr = rem - n * NBRS;
            p0 = tgt + b * DTGT;
            p1 = nodes + (b * NN + n) * DNODE;
            int di = s_next[(b * NN + n) * (NBRS + 1) + nbr];
            p2 = nodes + (b * NN + di) * DNODE;
            et = edge_type[(b * NN + n) * (NBRS + 1) + nbr];
        } else {
            int b = g / NN;
            int n = g - b * NN;
            p0 = tgt + b * DTGT;
            p1 = nodes + (b * NN + n) * DNODE;
        }
        const int span = KB1 * 8;                // K elems per thread
        for (int kk = 0; kk < span; ++kk) {
            int k = part * span + kk;
            float v;
            if (EDGE) {
                if      (k < 128) v = p0[k];
                else if (k < 256) v = p1[k - 128];
                else if (k < 384) v = p2[k - 256];
                else if (k == 384) v = (et == 1) ? 1.0f : 0.0f;
                else if (k == 385) v = (et == 2) ? 1.0f : 0.0f;
                else v = 0.0f;                   // K padding 386..415
            } else {
                v = (k < 128) ? p0[k] : p1[k - 128];
            }
            frag_store(sA, KB1, mtile, rfr, k, v);
        }
    }
    if (t < 64) sScore[t] = b3[0];
    __syncthreads();

    const int lane = t & 31, w = t >> 5;
    const int mtile = w >> 1, colgrp = w & 1;

    __builtin_prefetch(fragW2, 0, 0);            // warm L2 for layer 2 frags

    // ---- layer 1: K=KB1*32 -> 512, store hidden frags ---------------------
    gemm_layer<KB1, true, false>(sA, fragW1, b1, sH1, nullptr, nullptr,
                                 mtile, colgrp, lane);
    __syncthreads();

    // ---- layer 2: 512 -> 512, layer 3 (512 -> 1) folded into epilogue -----
    float sacc[8];
#pragma unroll
    for (int r = 0; r < 8; ++r) sacc[r] = 0.0f;
    gemm_layer<16, false, true>(sH1, fragW2, b2, nullptr, w3, sacc,
                                mtile, colgrp, lane);

    const int h = lane >> 4;
#pragma unroll
    for (int r = 0; r < 8; ++r)
        atomicAdd(&sScore[mtile * 16 + r + (h << 3)], sacc[r]);
    __syncthreads();

    if (t < 64) out_scores[blockIdx.x * 64 + t] = sScore[t];
}

// ---------------------------------------------------------------------------
// Masked 17-way softmax + log  (one thread per (b,n))
// ---------------------------------------------------------------------------
__global__ __launch_bounds__(256) void softmax_kernel(
    const float* __restrict__ edge_scores, const float* __restrict__ goal_scores,
    const int* __restrict__ edge_type, const int* __restrict__ node_masks,
    float* __restrict__ log_pi) {
    int idx = blockIdx.x * 256 + threadIdx.x;
    if (idx >= BB * NN) return;
    const int* et = edge_type + idx * (NBRS + 1);
    float logit[17];
#pragma unroll
    for (int j = 0; j < 16; ++j) {
        int e = et[j];
        float s  = edge_scores[idx * NBRS + j];
        float pe = (e == 1 || e == 2) ? s : 0.0f;
        logit[j] = (e != 0) ? pe : NEGV;
    }
    {
        int e = et[16];
        float g = (node_masks[idx] == 0) ? goal_scores[idx] : 0.0f;
        logit[16] = (e != 0) ? g : NEGV;
    }
    float m = logit[0];
#pragma unroll
    for (int j = 1; j < 17; ++j) m = fmaxf(m, logit[j]);
    float sum = 0.0f;
#pragma unroll
    for (int j = 0; j < 17; ++j) { logit[j] = expf(logit[j] - m); sum += logit[j]; }
    float inv = 1.0f / sum;
#pragma unroll
    for (int j = 0; j < 17; ++j)
        log_pi[idx * 17 + j] = logf(logit[j] * inv + 1e-5f);
}

// ---------------------------------------------------------------------------
// Multi-head attention + output assembly (tiny: <0.3% of FLOPs -> VALU).
// One block per batch element.
// ---------------------------------------------------------------------------
__global__ __launch_bounds__(256) void attention_kernel(
    const float* __restrict__ tgt, const float* __restrict__ nodes,
    const int* __restrict__ node_seq_gt,
    const float* __restrict__ qw, const float* __restrict__ qbi,
    const float* __restrict__ kw, const float* __restrict__ kbi,
    const float* __restrict__ vw, const float* __restrict__ vbi,
    const float* __restrict__ mqw, const float* __restrict__ mqb,
    const float* __restrict__ mkw, const float* __restrict__ mkb,
    const float* __restrict__ mvw, const float* __restrict__ mvb,
    const float* __restrict__ mow, const float* __restrict__ mob,
    float* __restrict__ out) {

    const int b = blockIdx.x, t = threadIdx.x;
    __shared__ float s_tgt[128];
    __shared__ float s_sel[HOR][128];
    __shared__ float s_k[HOR][EMB], s_v[HOR][EMB];
    __shared__ float s_kh[HOR][EMB], s_vh[HOR][EMB];
    __shared__ float s_q[EMB], s_qh[EMB];
    __shared__ float s_att[8][HOR];
    __shared__ float s_ctx[EMB], s_op[EMB];

    if (t < 128) s_tgt[t] = tgt[b * DTGT + t];
    for (int i = t; i < HOR * 128; i += 256) {
        int s = i >> 7, c = i & 127;
        int ns = node_seq_gt[b * HOR + s];
        float base = (ns < NN) ? nodes[(b * NN + ns) * DNODE + c] : 0.0f;
        int j = (c < 64) ? c : c - 64;
        float ang = (float)s * powf(10000.0f, -(float)j / 64.0f);
        s_sel[s][c] = base + ((c < 64) ? sinf(ang) : cosf(ang));
    }
    __syncthreads();

    {   // q / k / v embeddings (in-dim 128)
        float aq = qbi[t];
        for (int c = 0; c < 128; ++c) aq += s_tgt[c] * qw[c * EMB + t];
        s_q[t] = aq;
    }
    for (int i = t; i < HOR * EMB; i += 256) {
        int s = i >> 8, e = i & 255;
        float ak = kbi[e], av = vbi[e];
        for (int c = 0; c < 128; ++c) {
            float x = s_sel[s][c];
            ak += x * kw[c * EMB + e];
            av += x * vw[c * EMB + e];
        }
        s_k[s][e] = ak; s_v[s][e] = av;
    }
    __syncthreads();

    {   // head projections (in-dim 256)
        float aq = mqb[t];
        for (int c = 0; c < EMB; ++c) aq += s_q[c] * mqw[c * EMB + t];
        s_qh[t] = aq;
    }
    for (int i = t; i < HOR * EMB; i += 256) {
        int s = i >> 8, e = i & 255;
        float ak = mkb[e], av = mvb[e];
        for (int c = 0; c < EMB; ++c) {
            ak += s_k[s][c] * mkw[c * EMB + e];
            av += s_v[s][c] * mvw[c * EMB + e];
        }
        s_kh[s][e] = ak; s_vh[s][e] = av;
    }
    __syncthreads();

    if (t < 8 * HOR) {           // attention logits
        int h = t >> 4, s = t & 15;
        float a = 0.0f;
        for (int d = 0; d < 32; ++d) a += s_qh[h * 32 + d] * s_kh[s][h * 32 + d];
        a *= 0.17677669529663687f;               // 32^-0.5
        if (node_seq_gt[b * HOR + s] >= NN) a = NEGV;
        s_att[h][s] = a;
    }
    __syncthreads();
    if (t < 8) {                 // per-head softmax over HOR=16
        float m = -1e30f;
        for (int s = 0; s < HOR; ++s) m = fmaxf(m, s_att[t][s]);
        float sum = 0.0f;
        for (int s = 0; s < HOR; ++s) { float e = expf(s_att[t][s] - m); s_att[t][s] = e; sum += e; }
        float inv = 1.0f / sum;
        for (int s = 0; s < HOR; ++s) s_att[t][s] *= inv;
    }
    __syncthreads();
    {                            // context
        int h = t >> 5;
        float a = 0.0f;
        for (int s = 0; s < HOR; ++s) a += s_att[h][s] * s_vh[s][t];
        s_ctx[t] = a;
    }
    __syncthreads();
    {                            // output projection
        float a = mob[t];
        for (int c = 0; c < EMB; ++c) a += s_ctx[c] * mow[c * EMB + t];
        s_op[t] = a;
    }
    __syncthreads();
    for (int i = t; i < NSAMP * (DTGT + EMB); i += 256) {
        int c = i % (DTGT + EMB);
        out[b * NSAMP * (DTGT + EMB) + i] = (c < DTGT) ? s_tgt[c] : s_op[c - DTGT];
    }
}

// ---------------------------------------------------------------------------
// Launch
// ---------------------------------------------------------------------------
extern "C" void kernel_launch(void* const* d_in, const int* in_sizes, int n_in,
                              void* d_out, int out_size, void* d_ws, size_t ws_size,
                              hipStream_t stream) {
    (void)in_sizes; (void)n_in; (void)out_size; (void)ws_size;

    const float* tgt      = (const float*)d_in[0];
    const float* nodes    = (const float*)d_in[1];
    const float* pi_h1_w  = (const float*)d_in[4];
    const float* pi_h1_b  = (const float*)d_in[5];
    const float* pi_h2_w  = (const float*)d_in[6];
    const float* pi_h2_b  = (const float*)d_in[7];
    const float* pi_op_w  = (const float*)d_in[8];
    const float* pi_op_b  = (const float*)d_in[9];
    const float* pi_h1g_w = (const float*)d_in[10];
    const float* pi_h1g_b = (const float*)d_in[11];
    const float* pi_h2g_w = (const float*)d_in[12];
    const float* pi_h2g_b = (const float*)d_in[13];
    const float* pi_opg_w = (const float*)d_in[14];
    const float* pi_opg_b = (const float*)d_in[15];
    const float* q_emb_w  = (const float*)d_in[16];
    const float* q_emb_b  = (const float*)d_in[17];
    const float* k_emb_w  = (const float*)d_in[18];
    const float* k_emb_b  = (const float*)d_in[19];
    const float* v_emb_w  = (const float*)d_in[20];
    const float* v_emb_b  = (const float*)d_in[21];
    const float* mq_w     = (const float*)d_in[22];
    const float* mq_b     = (const float*)d_in[23];
    const float* mk_w     = (const float*)d_in[24];
    const float* mk_b     = (const float*)d_in[25];
    const float* mv_w     = (const float*)d_in[26];
    const float* mv_b     = (const float*)d_in[27];
    const float* mo_w     = (const float*)d_in[28];
    const float* mo_b     = (const float*)d_in[29];
    const int* node_masks  = (const int*)d_in[30];
    const int* s_next      = (const int*)d_in[31];
    const int* edge_type   = (const int*)d_in[32];
    const int* node_seq_gt = (const int*)d_in[33];

    float* out = (float*)d_out;

    // workspace carve-up (all offsets 32B-aligned)
    _Float16* fragW1e = (_Float16*)d_ws;                 // 13*16384 halves
    _Float16* fragW2e = fragW1e + 13 * 16384;            // 16*16384
    _Float16* fragW1g = fragW2e + 16 * 16384;            //  8*16384
    _Float16* fragW2g = fragW1g + 8 * 16384;             // 16*16384
    float* edge_scores = (float*)(fragW2g + 16 * 16384); // 131072 f32
    float* goal_scores = edge_scores + BB * NN * NBRS;   //   8192 f32

    // weight re-layout into WMMA B-fragment order
    prep_w_kernel<<<(13 * 16384) / 256, 256, 0, stream>>>(pi_h1_w,  fragW1e, 13, 386);
    prep_w_kernel<<<(16 * 16384) / 256, 256, 0, stream>>>(pi_h2_w,  fragW2e, 16, 512);
    prep_w_kernel<<<( 8 * 16384) / 256, 256, 0, stream>>>(pi_h1g_w, fragW1g,  8, 256);
    prep_w_kernel<<<(16 * 16384) / 256, 256, 0, stream>>>(pi_h2g_w, fragW2g, 16, 512);

    // edge MLP: 131072 rows / 64 per block
    mlp3_kernel<13, true><<<(BB * NN * NBRS) / 64, 256, 0, stream>>>(
        tgt, nodes, s_next, edge_type, fragW1e, fragW2e,
        pi_h1_b, pi_h2_b, pi_op_w, pi_op_b, edge_scores);

    // goal MLP: 8192 rows / 64 per block
    mlp3_kernel<8, false><<<(BB * NN) / 64, 256, 0, stream>>>(
        tgt, nodes, nullptr, nullptr, fragW1g, fragW2g,
        pi_h1g_b, pi_h2g_b, pi_opg_w, pi_opg_b, goal_scores);

    // masked softmax + log  -> log_pi lives after agg_enc in d_out
    softmax_kernel<<<(BB * NN) / 256, 256, 0, stream>>>(
        edge_scores, goal_scores, edge_type, node_masks, out + AGG_ELEMS);

    // attention + agg_enc broadcast
    attention_kernel<<<BB, 256, 0, stream>>>(
        tgt, nodes, node_seq_gt,
        q_emb_w, q_emb_b, k_emb_w, k_emb_b, v_emb_w, v_emb_b,
        mq_w, mq_b, mk_w, mk_b, mv_w, mv_b, mo_w, mo_b, out);
}
